// DMoLELinear_80908593922352
// MI455X (gfx1250) — matrix operand order repaired and checked by reading, not compiled, not run on hardware
//
#include <hip/hip_runtime.h>
#include <hip/hip_bf16.h>

// ---------------------------------------------------------------------------
// DMoLE linear:  out = x @ W^T + b + s * sum_e m_e (x A_e^T) B_e^T
// Collapsed to:  W_eff = W + s * sum_e m_e B_e A_e   (rank-128 update, tiny)
//                out   = x @ W_eff^T + b             (8192x2048x2048 GEMM)
// GEMM: bf16x3 split precision on v_wmma_f32_16x16x32_bf16, double-buffered
// LDS (BK=64), W_eff tiles staged with GLOBAL_LOAD_ASYNC_TO_LDS_B128
// (ASYNCcnt), X split-converted with v_perm_b32 packing.
// ---------------------------------------------------------------------------

#define B_DIM   4
#define S_DIM   2048
#define M_DIM   (B_DIM * S_DIM)   // 8192 token rows
#define N_DIM   2048              // D_OUT
#define K_DIM   2048              // D_IN
#define E_DIM   8
#define R_DIM   16
#define SCALING 1.0f

typedef __attribute__((ext_vector_type(16))) __bf16        v16bf;
typedef __attribute__((ext_vector_type(8)))  float         v8f;
typedef __attribute__((ext_vector_type(4)))  unsigned int  v4u;
typedef __attribute__((ext_vector_type(2)))  unsigned int  v2u;

union FragBF { v16bf v; v4u q[2]; };

__device__ __forceinline__ unsigned bf16rne_u32(unsigned u) {
    return u + 0x7FFFu + ((u >> 16) & 1u);    // round-to-nearest-even in bits
}
__device__ __forceinline__ unsigned short f32_to_bf16_rne(float f) {
    return (unsigned short)(bf16rne_u32(__builtin_bit_cast(unsigned, f)) >> 16);
}
__device__ __forceinline__ float bf16_to_f32(unsigned short h) {
    unsigned int u = ((unsigned int)h) << 16;
    return __builtin_bit_cast(float, u);
}
// Split two f32 into packed bf16 hi pair + bf16 lo pair using v_perm_b32.
__device__ __forceinline__ void split2(float x, float y,
                                       unsigned& hp, unsigned& lp) {
    const unsigned ux = __builtin_bit_cast(unsigned, x);
    const unsigned uy = __builtin_bit_cast(unsigned, y);
    const unsigned rx = bf16rne_u32(ux);
    const unsigned ry = bf16rne_u32(uy);
    hp = __builtin_amdgcn_perm(ry, rx, 0x07060302u);   // {ry[31:16],rx[31:16]}
    const float lx = x - __builtin_bit_cast(float, rx & 0xFFFF0000u);
    const float ly = y - __builtin_bit_cast(float, ry & 0xFFFF0000u);
    const unsigned sx = bf16rne_u32(__builtin_bit_cast(unsigned, lx));
    const unsigned sy = bf16rne_u32(__builtin_bit_cast(unsigned, ly));
    lp = __builtin_amdgcn_perm(sy, sx, 0x07060302u);
}
// Flat LDS addresses keep the wave-relative LDS byte offset in addr[31:0].
__device__ __forceinline__ unsigned lds_off32(const void* p) {
    return (unsigned)(unsigned long long)p;
}
__device__ __forceinline__ void wait_async0() {
#if __has_builtin(__builtin_amdgcn_s_wait_asynccnt)
    __builtin_amdgcn_s_wait_asynccnt(0);
#else
    asm volatile("s_wait_asynccnt 0x0" ::: "memory");
#endif
}
// Async copy 16B: LDS[ldsoff] = MEM[saddr + voff]  (GVS mode, ASYNCcnt++)
__device__ __forceinline__ void async_b128(unsigned ldsoff, unsigned voff,
                                           unsigned long long saddr) {
    asm volatile("global_load_async_to_lds_b128 %0, %1, %2"
                 :: "v"(ldsoff), "v"(voff), "s"(saddr) : "memory");
}

// ---------------------------------------------------------------------------
// Kernel 1: fold masked LoRA experts into W, emit split bf16 hi/lo planes.
// ---------------------------------------------------------------------------
__global__ __launch_bounds__(256) void fold_lora_kernel(
    const float* __restrict__ W,          // [N,K]
    const float* __restrict__ loraA,      // [E,R,K]
    const float* __restrict__ loraB,      // [E,N,R]
    const unsigned char* __restrict__ mask,
    unsigned short* __restrict__ Wh,      // [N,K] bf16 hi
    unsigned short* __restrict__ Wl)      // [N,K] bf16 lo
{
    const int o = blockIdx.x >> 3;                           // uniform per block
    const int d = ((blockIdx.x & 7) << 8) + threadIdx.x;     // coalesced

    float acc = 0.0f;
    #pragma unroll
    for (int e = 0; e < E_DIM; ++e) {
        if (!mask[e]) continue;                              // uniform branch
        const float* Ae = loraA + ((size_t)e * R_DIM) * K_DIM + d;
        const float* Be = loraB + ((size_t)e * N_DIM + o) * R_DIM;
        #pragma unroll
        for (int r = 0; r < R_DIM; ++r)
            acc += Be[r] * Ae[(size_t)r * K_DIM];
    }
    const size_t idx = (size_t)o * K_DIM + d;
    const float w = W[idx] + SCALING * acc;
    const unsigned short hi = f32_to_bf16_rne(w);
    Wh[idx] = hi;
    Wl[idx] = f32_to_bf16_rne(w - bf16_to_f32(hi));
}

// ---------------------------------------------------------------------------
// Kernel 2: C = X @ Weff^T + b  via bf16x3 WMMA, double-buffered pipeline.
// Block tile 128x128, BK=64 (2 WMMA k-steps per stage). 8 waves: 4(M) x 2(N);
// wave tile 32x64 = 2x4 WMMA tiles; 3 WMMAs (hh,hl,lh) per tile per k-step.
// ---------------------------------------------------------------------------
#define BM   128
#define BN   128
#define BK   64
#define LDSS 72   // padded LDS row stride in bf16 elems (144 B; 16B-aligned)

__global__ __launch_bounds__(256) void dmole_gemm_kernel(
    const float* __restrict__ X,           // [M,K] f32
    const unsigned short* __restrict__ Wh, // [N,K] bf16 hi
    const unsigned short* __restrict__ Wl, // [N,K] bf16 lo
    const float* __restrict__ bias,        // [N]
    float* __restrict__ out)               // [M,N] f32
{
    __shared__ unsigned short sXh[2][BM * LDSS];
    __shared__ unsigned short sXl[2][BM * LDSS];
    __shared__ unsigned short sWh[2][BN * LDSS];
    __shared__ unsigned short sWl[2][BN * LDSS];

    const int t    = threadIdx.x;
    const int lane = t & 31;
    const int wave = t >> 5;
    const int wm   = wave >> 1;        // 0..3 -> 32-row slabs of M
    const int wn   = wave & 1;         // 0..1 -> 64-col slabs of N
    const int h    = lane >> 4;        // lane half (WMMA K-phase select)
    const int l16  = lane & 15;

    const int m0 = blockIdx.y * BM;
    const int n0 = blockIdx.x * BN;

    const unsigned long long whp = (unsigned long long)Wh;
    const unsigned long long wlp = (unsigned long long)Wl;

    v8f acc[2][4] = {};

    // ---- W_eff tile staging: async global->LDS, no VGPR transit ----
    auto issueW = [&](int kt, int buf) {
        #pragma unroll
        for (int it = 0; it < 4; ++it) {
            const int idx = t + (it << 8);      // 0..1023
            const int row = idx >> 3;           // 128 rows x 8 chunks
            const int c8  = (idx & 7) << 3;     // bf16 col start (16B chunks)
            const unsigned goff =
                (unsigned)(((n0 + row) * K_DIM + kt * BK + c8) * 2);
            async_b128(lds_off32(&sWh[buf][row * LDSS + c8]), goff, whp);
            async_b128(lds_off32(&sWl[buf][row * LDSS + c8]), goff, wlp);
        }
    };
    // ---- X tile: global f32 -> regs ----
    auto loadX = [&](int kt, float4* xr) {
        #pragma unroll
        for (int it = 0; it < 8; ++it) {
            const int idx = t + (it << 8);      // 0..2047
            const int row = idx >> 4;           // 128 rows x 16 float4
            const int c4  = (idx & 15) << 2;
            xr[it] = *(const float4*)(X + (size_t)(m0 + row) * K_DIM
                                        + kt * BK + c4);
        }
    };
    // ---- X tile: split f32 -> hi/lo bf16 (v_perm pack), store to LDS ----
    auto storeX = [&](const float4* xr, int buf) {
        #pragma unroll
        for (int it = 0; it < 8; ++it) {
            const int idx = t + (it << 8);
            const int row = idx >> 4;
            const int c4  = (idx & 15) << 2;
            const float4 xv = xr[it];
            unsigned h0, h1, l0, l1;
            split2(xv.x, xv.y, h0, l0);
            split2(xv.z, xv.w, h1, l1);
            v2u hp; hp.x = h0; hp.y = h1;
            v2u lp; lp.x = l0; lp.y = l1;
            *(v2u*)&sXh[buf][row * LDSS + c4] = hp;  // 8B aligned
            *(v2u*)&sXl[buf][row * LDSS + c4] = lp;
        }
    };
    // ---- fragment loads + 48 WMMAs from LDS buffer `buf` ----
    auto compute = [&](int buf) {
        #pragma unroll
        for (int kk = 0; kk < 2; ++kk) {
            const int ko = kk * 32;             // k-step within the stage
            FragBF bh[4], bl[4];
            #pragma unroll
            for (int j = 0; j < 4; ++j) {       // B frag: 32B contiguous @16h
                const int rb = wn * 64 + j * 16 + l16;
                const unsigned short* ph = &sWh[buf][rb * LDSS + ko + h * 16];
                const unsigned short* pl = &sWl[buf][rb * LDSS + ko + h * 16];
                bh[j].q[0] = *(const v4u*)ph;  bh[j].q[1] = *(const v4u*)(ph + 8);
                bl[j].q[0] = *(const v4u*)pl;  bl[j].q[1] = *(const v4u*)(pl + 8);
            }
            #pragma unroll
            for (int i = 0; i < 2; ++i) {       // A frag: 32B contiguous @16h
                const int ra = wm * 32 + i * 16 + l16;
                FragBF ah, al;
                ah.q[0] = *(const v4u*)&sXh[buf][ra * LDSS + ko + h * 16];
                ah.q[1] = *(const v4u*)&sXh[buf][ra * LDSS + ko + h * 16 + 8];
                al.q[0] = *(const v4u*)&sXl[buf][ra * LDSS + ko + h * 16];
                al.q[1] = *(const v4u*)&sXl[buf][ra * LDSS + ko + h * 16 + 8];
                #pragma unroll
                for (int j = 0; j < 4; ++j) {
                    acc[i][j] = __builtin_amdgcn_wmma_f32_16x16x32_bf16(
                        false, ah.v, false, bh[j].v, (short)0, acc[i][j], false, false);
                    acc[i][j] = __builtin_amdgcn_wmma_f32_16x16x32_bf16(
                        false, ah.v, false, bl[j].v, (short)0, acc[i][j], false, false);
                    acc[i][j] = __builtin_amdgcn_wmma_f32_16x16x32_bf16(
                        false, al.v, false, bh[j].v, (short)0, acc[i][j], false, false);
                }
            }
        }
    };

    // ---- pipeline prologue: stage tile 0 into buffer 0 ----
    float4 xr[8];
    issueW(0, 0);
    loadX(0, xr);
    storeX(xr, 0);
    wait_async0();
    __syncthreads();

    const int KT = K_DIM / BK;                  // 32 iterations
    for (int kt = 0; kt < KT; ++kt) {
        const int cur  = kt & 1;
        const int nxt  = cur ^ 1;
        const bool more = (kt + 1) < KT;
        if (more) {
            issueW(kt + 1, nxt);   // async -> LDS[nxt], overlaps with WMMA
            loadX(kt + 1, xr);     // long-latency global loads issue early
        }
        compute(cur);
        if (more) storeX(xr, nxt); // convert after compute (hides load wait)
        wait_async0();             // own wave's async W tiles landed
        __syncthreads();           // all waves' tiles visible
    }

    // ---- epilogue: C/D layout -> m = tile_m + v + 8h, n = tile_n + l16 ----
    #pragma unroll
    for (int i = 0; i < 2; ++i) {
        #pragma unroll
        for (int j = 0; j < 4; ++j) {
            const int n  = n0 + wn * 64 + j * 16 + l16;
            const float bb = bias[n];
            const int mb = m0 + wm * 32 + i * 16 + h * 8;
            #pragma unroll
            for (int v = 0; v < 8; ++v)
                out[(size_t)(mb + v) * N_DIM + n] = acc[i][j][v] + bb;
        }
    }
}

// ---------------------------------------------------------------------------
extern "C" void kernel_launch(void* const* d_in, const int* in_sizes, int n_in,
                              void* d_out, int out_size, void* d_ws, size_t ws_size,
                              hipStream_t stream) {
    const float*         X     = (const float*)d_in[0];          // [4,2048,2048]
    const float*         W     = (const float*)d_in[1];          // [2048,2048]
    const float*         bias  = (const float*)d_in[2];          // [2048]
    const float*         loraA = (const float*)d_in[3];          // [8,16,2048]
    const float*         loraB = (const float*)d_in[4];          // [8,2048,16]
    const unsigned char* mask  = (const unsigned char*)d_in[5];  // [8] bool
    float*               out   = (float*)d_out;                  // [4,2048,2048]

    unsigned short* Wh = (unsigned short*)d_ws;                  // 8 MB
    unsigned short* Wl = Wh + (size_t)N_DIM * K_DIM;             // 8 MB

    // Fold LoRA rank-128 update into W, split to bf16 hi/lo.
    fold_lora_kernel<<<dim3((N_DIM * K_DIM) / 256), dim3(256), 0, stream>>>(
        W, loraA, loraB, mask, Wh, Wl);

    // Main GEMM with fused bias.
    dmole_gemm_kernel<<<dim3(N_DIM / BN, M_DIM / BM), dim3(256), 0, stream>>>(
        X, Wh, Wl, bias, out);
}
